// SwinTransformerBlock_35966056137269
// MI455X (gfx1250) — matrix operand order repaired and checked by reading, not compiled, not run on hardware
//
#include <hip/hip_runtime.h>
#include <hip/hip_bf16.h>
#include <math.h>

typedef __attribute__((ext_vector_type(16))) _Float16 v16h;
typedef __attribute__((ext_vector_type(8)))  _Float16 v8h;
typedef __attribute__((ext_vector_type(8)))  float    v8f;
typedef __attribute__((ext_vector_type(4)))  _Float16 v4h;

namespace {

constexpr float kQScale = 0.17677669529663689f; // 1/sqrt(32)

// A-fragment (16x32 f16, MxK) from row-major LDS tile. ISA 7.12.2:
// lanes 0-15: M=lane, K-halves {0..7,16..23}; lanes 16-31: M=lane-16, K {8..15,24..31}.
// Each lane's data = two contiguous 16B chunks -> 2x ds_load_b128.
__device__ __forceinline__ v16h lds_afrag(const _Float16* base, int stride, int lane) {
  int m  = lane & 15;
  int kb = (lane & 16) ? 8 : 0;
  const _Float16* p = base + m * stride + kb;
  v8h lo = *(const v8h*)(p);
  v8h hi = *(const v8h*)(p + 16);
  return __builtin_shufflevector(lo, hi, 0, 1, 2, 3, 4, 5, 6, 7,
                                 8, 9, 10, 11, 12, 13, 14, 15);
}

// B-fragment (32x16 f16, KxN) from row-major LDS tile: lane holds column n=lane&15,
// K {0..7,16..23} (lanes 0-15) / {8..15,24..31} (lanes 16-31), matching A.
__device__ __forceinline__ v16h lds_bfrag(const _Float16* base, int stride, int lane) {
  int kb = (lane & 16) ? 8 : 0;
  const _Float16* p = base + (lane & 15);
  v16h r;
#pragma unroll
  for (int s = 0; s < 8; ++s) {
    r[s] = p[(kb + s) * stride];
    r[s + 8] = p[(16 + kb + s) * stride];
  }
  return r;
}

__device__ __forceinline__ int regid3(int a) { return a < 49 ? 0 : (a < 53 ? 1 : 2); }

} // namespace

// ---------------- init kernels ----------------

__global__ __launch_bounds__(256) void convw_kernel(
    const float* __restrict__ qkv_w, const float* __restrict__ proj_w,
    const float* __restrict__ fc1_w, const float* __restrict__ fc2_w,
    _Float16* __restrict__ Wq, _Float16* __restrict__ Wp,
    _Float16* __restrict__ W1, _Float16* __restrict__ W2) {
  int idx = blockIdx.x * 256 + threadIdx.x;
  if (idx < 196608)       Wq[idx]          = (_Float16)qkv_w[idx];
  else if (idx < 262144)  Wp[idx - 196608] = (_Float16)proj_w[idx - 196608];
  else if (idx < 524288)  W1[idx - 262144] = (_Float16)fc1_w[idx - 262144];
  else if (idx < 786432)  W2[idx - 524288] = (_Float16)fc2_w[idx - 524288];
}

// Combined, padded attention bias: cb[head][wloc][i(64)][j(64)] =
//   j>=49 ? -1e30 : (i>=49 ? 0 : rpb(head,i,j) + mask(wloc,i,j))
__global__ __launch_bounds__(256) void cbias_kernel(const float* __restrict__ table,
                                                    float* __restrict__ cb) {
  int idx = blockIdx.x * 256 + threadIdx.x;   // 8*64*64*64 = 2097152
  int j = idx & 63;
  int i = (idx >> 6) & 63;
  int wloc = (idx >> 12) & 63;
  int head = idx >> 18;
  float v;
  if (j >= 49) {
    v = -1e30f;
  } else if (i >= 49) {
    v = 0.f;
  } else {
    int dx = (i % 7) - (j % 7) + 6;   // meshgrid 'xy': x = col
    int dy = (i / 7) - (j / 7) + 6;
    float r = table[(dx * 13 + dy) * 8 + head];
    int wr = wloc >> 3, wc = wloc & 7;
    int id1 = regid3(wr * 7 + i / 7) * 3 + regid3(wc * 7 + i % 7);
    int id2 = regid3(wr * 7 + j / 7) * 3 + regid3(wc * 7 + j % 7);
    v = r + ((id1 != id2) ? -100.f : 0.f);
  }
  cb[idx] = v;
}

// ---------------- LayerNorm (+ optional shift/window gather), f32 -> f16 ----------------

__global__ __launch_bounds__(256) void ln_kernel(const float* __restrict__ src,
                                                 const float* __restrict__ scale,
                                                 const float* __restrict__ bias,
                                                 _Float16* __restrict__ dst,
                                                 int shifted) {
  int lane = threadIdx.x & 31;
  int wave = threadIdx.x >> 5;
  int t = blockIdx.x * 8 + wave;   // destination row 0..100351
  size_t srcTok;
  if (shifted) {
    int win = t / 49, n = t % 49;
    int b = win >> 6, wl = win & 63;
    int wr = wl >> 3, wc = wl & 7;
    int r = n / 7, cc = n % 7;
    int h = (wr * 7 + r + 3) % 56;   // roll by -3: rolled[i] = x[(i+3)%56]
    int w = (wc * 7 + cc + 3) % 56;
    srcTok = (size_t)b * 3136 + h * 56 + w;
  } else {
    srcTok = (size_t)t;
  }
  const float* p = src + srcTok * 256;
  float v[8];
  float s = 0.f;
#pragma unroll
  for (int j = 0; j < 8; ++j) { v[j] = p[lane + 32 * j]; s += v[j]; }
#pragma unroll
  for (int off = 16; off >= 1; off >>= 1) s += __shfl_xor(s, off);
  float mu = s * (1.f / 256.f);
  float q = 0.f;
#pragma unroll
  for (int j = 0; j < 8; ++j) { float d = v[j] - mu; q += d * d; }
#pragma unroll
  for (int off = 16; off >= 1; off >>= 1) q += __shfl_xor(q, off);
  float rstd = rsqrtf(q * (1.f / 256.f) + 1e-6f);
  _Float16* o = dst + (size_t)t * 256;
#pragma unroll
  for (int j = 0; j < 8; ++j) {
    int c = lane + 32 * j;
    o[c] = (_Float16)((v[j] - mu) * rstd * scale[c] + bias[c]);
  }
}

// ---------------- tiled f16 GEMM with fused epilogues ----------------
// Block tile 128(M) x 128(N), 256 threads = 8 waves; each wave owns 16 M-rows and
// all 128 N-cols (8 accumulators -> 8 v_wmma per k-step).
// EPI: 0=QKV(scale q, per-head f16 scatter)  1=proj(window-reverse scatter + residual)
//      2=fc1(GELU, f16)  3=fc2(residual RMW on f32 out)

template <int EPI>
__global__ __launch_bounds__(256) void gemm_kernel(
    const _Float16* __restrict__ A, const _Float16* __restrict__ Bw,
    const float* __restrict__ bias, float* __restrict__ outF,
    _Float16* __restrict__ outH, const float* __restrict__ x0,
    int M, int N, int K) {
  __shared__ __align__(16) _Float16 As[128 * 32];
  __shared__ __align__(16) _Float16 Bs[32 * 128];
  int tid = threadIdx.x, lane = tid & 31, wave = tid >> 5;
  int m0 = blockIdx.y * 128, n0 = blockIdx.x * 128;
  v8f acc[8] = {};
  for (int k0 = 0; k0 < K; k0 += 32) {
#pragma unroll
    for (int i = 0; i < 4; ++i) {              // A tile: 128x32 halves
      int c = tid + i * 256;
      int row = c >> 3, c4 = c & 7;
      *(v4h*)&As[row * 32 + c4 * 4] =
          *(const v4h*)&A[(size_t)(m0 + row) * K + k0 + c4 * 4];
    }
#pragma unroll
    for (int i = 0; i < 4; ++i) {              // B tile: 32x128 halves
      int c = tid + i * 256;
      int row = c >> 5, c4 = c & 31;
      *(v4h*)&Bs[row * 128 + c4 * 4] =
          *(const v4h*)&Bw[(size_t)(k0 + row) * N + n0 + c4 * 4];
    }
    if (k0 + 32 < K)                            // global_prefetch_b8 for next A slab
      __builtin_prefetch(&A[(size_t)(m0 + (tid >> 1)) * K + k0 + 32], 0, 0);
    __syncthreads();
    v16h af = lds_afrag(&As[wave * 16 * 32], 32, lane);
#pragma unroll
    for (int nt = 0; nt < 8; ++nt) {
      v16h bf = lds_bfrag(&Bs[nt * 16], 128, lane);
      acc[nt] = __builtin_amdgcn_wmma_f32_16x16x32_f16(false, af, false, bf,
                                                       (short)0, acc[nt], false, false);
    }
    __syncthreads();
  }
  int hi8 = (lane & 16) ? 8 : 0;
  int nl = lane & 15;
#pragma unroll
  for (int nt = 0; nt < 8; ++nt) {
#pragma unroll
    for (int v = 0; v < 8; ++v) {
      int m = m0 + wave * 16 + hi8 + v;
      int n = n0 + nt * 16 + nl;
      float val = acc[nt][v] + bias[n];
      if (EPI == 0) {
        if (n < 256) val *= kQScale;
        int win = m / 49, nn = m % 49;
        int which = n >> 8, head = (n >> 5) & 7, d = n & 31;
        outH[((((size_t)win * 3 + which) * 8 + head) * 49 + nn) * 32 + d] = (_Float16)val;
      } else if (EPI == 1) {
        int win = m / 49, nn = m % 49;
        int b = win >> 6, wl = win & 63;
        int wr = wl >> 3, wc = wl & 7;
        int r = nn / 7, cc = nn % 7;
        int h = (wr * 7 + r + 3) % 56;          // reverse shift: roll(+3)
        int w = (wc * 7 + cc + 3) % 56;
        size_t idx = ((size_t)b * 3136 + h * 56 + w) * 256 + n;
        outF[idx] = x0[idx] + val;
      } else if (EPI == 2) {
        float u = val;
        float t3 = tanhf(0.7978845608028654f * (u + 0.044715f * u * u * u));
        outH[(size_t)m * N + n] = (_Float16)(0.5f * u * (1.f + t3));
      } else {
        size_t idx = (size_t)m * 256 + n;
        outF[idx] = outF[idx] + val;
      }
    }
  }
}

// ---------------- windowed attention: one block per (window, head) ----------------

__global__ __launch_bounds__(128) void attn_kernel(const _Float16* __restrict__ qkv,
                                                   const float* __restrict__ cb,
                                                   _Float16* __restrict__ outA) {
  __shared__ __align__(16) _Float16 Qs[64 * 32];   // Q rows, zero-padded
  __shared__ __align__(16) _Float16 Kt[32 * 64];   // K transposed: [k][j], j-pad zero
  __shared__ __align__(16) _Float16 Vs[64 * 32];   // V rows, zero-padded
  __shared__ __align__(16) _Float16 Ps[64 * 64];   // softmax(P) f16
  int win = blockIdx.x >> 3;
  int head = blockIdx.x & 7;
  int tid = threadIdx.x, lane = tid & 31, wave = tid >> 5;
  const _Float16* qb = qkv + ((size_t)win * 24 + head) * 1568;   // [win][3][8][49][32]
  const _Float16* kp = qb + (size_t)8 * 1568;
  const _Float16* vp = qb + (size_t)16 * 1568;
  for (int i = tid; i < 512; i += 128) {
    v4h z = {};
    *(v4h*)&Qs[i * 4] = (i < 392) ? *(const v4h*)&qb[i * 4] : z;
    *(v4h*)&Vs[i * 4] = (i < 392) ? *(const v4h*)&vp[i * 4] : z;
    // K transpose: i = j*8 + k-chunk; write 4 halves down a column of Kt
    int j = i >> 3, k4 = (i & 7) * 4;
    v4h kv = z;
    if (j < 49) kv = *(const v4h*)&kp[j * 32 + k4];
#pragma unroll
    for (int t = 0; t < 4; ++t) Kt[(k4 + t) * 64 + j] = kv[t];
  }
  __syncthreads();

  // S = (scaled Q) @ K^T : per wave 16x64, K-dim = 32 = one WMMA step
  v16h aq = lds_afrag(&Qs[wave * 16 * 32], 32, lane);
  v8f s[4] = {};
#pragma unroll
  for (int nt = 0; nt < 4; ++nt) {
    v16h bf = lds_bfrag(&Kt[nt * 16], 64, lane);   // B[k][n] = Kt[k][j]
    s[nt] = __builtin_amdgcn_wmma_f32_16x16x32_f16(false, aq, false, bf, (short)0,
                                                   s[nt], false, false);
  }

  int hi8 = (lane & 16) ? 8 : 0;
  int nl = lane & 15;
  int wloc = win & 63;
  // branchless combined bias (rpb + shift mask + j-padding baked in)
  const float* cbp = cb + (((size_t)head * 64 + wloc) << 12);
#pragma unroll
  for (int nt = 0; nt < 4; ++nt)
#pragma unroll
    for (int v = 0; v < 8; ++v)
      s[nt][v] += cbp[(wave * 16 + hi8 + v) * 64 + nt * 16 + nl];

  // softmax along j: each row lives in lanes (l&15) x 4 fragments.
  // xor masks {1,2,4,8} never cross a 16-lane group -> full-width shuffles are safe
  // (avoids the width-16 clamp cndmask chains).
#pragma unroll
  for (int v = 0; v < 8; ++v) {
    float mx = fmaxf(fmaxf(s[0][v], s[1][v]), fmaxf(s[2][v], s[3][v]));
#pragma unroll
    for (int off = 8; off >= 1; off >>= 1) mx = fmaxf(mx, __shfl_xor(mx, off));
    float sum = 0.f;
#pragma unroll
    for (int nt = 0; nt < 4; ++nt) { float e = __expf(s[nt][v] - mx); s[nt][v] = e; sum += e; }
#pragma unroll
    for (int off = 8; off >= 1; off >>= 1) sum += __shfl_xor(sum, off);
    float inv = 1.f / sum;
#pragma unroll
    for (int nt = 0; nt < 4; ++nt) s[nt][v] *= inv;
  }
#pragma unroll
  for (int nt = 0; nt < 4; ++nt)
#pragma unroll
    for (int v = 0; v < 8; ++v)
      Ps[(wave * 16 + hi8 + v) * 64 + nt * 16 + nl] = (_Float16)s[nt][v];
  __syncthreads();

  // O = P @ V : per wave 16x32, K-dim 64 (padded cols are exact zeros) = 2 steps
  v8f o[2] = {};
#pragma unroll
  for (int ks = 0; ks < 2; ++ks) {
    v16h ap = lds_afrag(&Ps[wave * 16 * 64 + ks * 32], 64, lane);
#pragma unroll
    for (int nt = 0; nt < 2; ++nt) {
      v16h bf = lds_bfrag(&Vs[(ks * 32) * 32 + nt * 16], 32, lane);
      o[nt] = __builtin_amdgcn_wmma_f32_16x16x32_f16(false, ap, false, bf, (short)0,
                                                     o[nt], false, false);
    }
  }
#pragma unroll
  for (int nt = 0; nt < 2; ++nt)
#pragma unroll
    for (int v = 0; v < 8; ++v) {
      int i = wave * 16 + hi8 + v;
      if (i < 49) {
        int d = nt * 16 + nl;
        outA[((size_t)win * 49 + i) * 256 + head * 32 + d] = (_Float16)o[nt][v];
      }
    }
}

// ---------------- launch ----------------

extern "C" void kernel_launch(void* const* d_in, const int* in_sizes, int n_in,
                              void* d_out, int out_size, void* d_ws, size_t ws_size,
                              hipStream_t stream) {
  (void)in_sizes; (void)n_in; (void)out_size; (void)ws_size;
  const float* x      = (const float*)d_in[0];
  const float* ln1_s  = (const float*)d_in[1];
  const float* ln1_b  = (const float*)d_in[2];
  const float* qkv_w  = (const float*)d_in[3];
  const float* qkv_b  = (const float*)d_in[4];
  const float* proj_w = (const float*)d_in[5];
  const float* proj_b = (const float*)d_in[6];
  const float* btab   = (const float*)d_in[7];
  const float* ln2_s  = (const float*)d_in[8];
  const float* ln2_b  = (const float*)d_in[9];
  const float* fc1_w  = (const float*)d_in[10];
  const float* fc1_b  = (const float*)d_in[11];
  const float* fc2_w  = (const float*)d_in[12];
  const float* fc2_b  = (const float*)d_in[13];
  float* out = (float*)d_out;

  char* ws = (char*)d_ws;
  size_t off = 0;
  auto carve = [&](size_t bytes) {
    char* p = ws + off;
    off = (off + bytes + 255) & ~(size_t)255;
    return p;
  };
  _Float16* Wq   = (_Float16*)carve((size_t)196608 * 2);
  _Float16* Wp   = (_Float16*)carve((size_t)65536 * 2);
  _Float16* W1   = (_Float16*)carve((size_t)262144 * 2);
  _Float16* W2   = (_Float16*)carve((size_t)262144 * 2);
  float*    cb   = (float*)carve((size_t)8 * 64 * 64 * 64 * 4);   // combined bias
  _Float16* Abuf = (_Float16*)carve((size_t)100352 * 256 * 2);    // win / attnout / ln2out
  _Float16* QKV  = (_Float16*)carve((size_t)100352 * 768 * 2);
  _Float16* F1   = (_Float16*)carve((size_t)100352 * 1024 * 2);

  convw_kernel<<<3072, 256, 0, stream>>>(qkv_w, proj_w, fc1_w, fc2_w, Wq, Wp, W1, W2);
  cbias_kernel<<<8192, 256, 0, stream>>>(btab, cb);

  // LN1 + shift + window partition -> f16
  ln_kernel<<<12544, 256, 0, stream>>>(x, ln1_s, ln1_b, Abuf, 1);
  // QKV gemm (q pre-scaled), per-head scatter
  gemm_kernel<0><<<dim3(6, 784), 256, 0, stream>>>(Abuf, Wq, qkv_b, nullptr, QKV, nullptr,
                                                   100352, 768, 256);
  // attention
  attn_kernel<<<16384, 128, 0, stream>>>(QKV, cb, Abuf);
  // proj + window reverse + reverse shift + residual -> out
  gemm_kernel<1><<<dim3(2, 784), 256, 0, stream>>>(Abuf, Wp, proj_b, out, nullptr, x,
                                                   100352, 256, 256);
  // LN2 -> f16
  ln_kernel<<<12544, 256, 0, stream>>>(out, ln2_s, ln2_b, Abuf, 0);
  // fc1 + GELU
  gemm_kernel<2><<<dim3(8, 784), 256, 0, stream>>>(Abuf, W1, fc1_b, nullptr, F1, nullptr,
                                                   100352, 1024, 256);
  // fc2 + residual (RMW on out)
  gemm_kernel<3><<<dim3(2, 784), 256, 0, stream>>>(F1, W2, fc2_b, out, nullptr, nullptr,
                                                   100352, 256, 1024);
}
